// SwinAttentionFusionAvg_36953898614925
// MI455X (gfx1250) — matrix operand order repaired and checked by reading, not compiled, not run on hardware
//
#include <hip/hip_runtime.h>
#include <hip/hip_bf16.h>
#include <math.h>

typedef __bf16 bf16_t;
typedef __attribute__((ext_vector_type(16))) __bf16 v16bf;
typedef __attribute__((ext_vector_type(8)))  float  v8f;

struct U16B { unsigned int w[4]; };           // POD 16-byte chunk for LDS vector copies
union FragU { v16bf v; bf16_t h[16]; U16B q[2]; };

__device__ __forceinline__ bf16_t f2bf(float x) { return (bf16_t)x; }

__device__ __forceinline__ v8f v8f_zero() {
  v8f z;
#pragma unroll
  for (int i = 0; i < 8; ++i) z[i] = 0.f;
  return z;
}

// ---- CDNA5 async global->LDS staging (ASYNCcnt path), with safe fallback ----
#if defined(__HIP_DEVICE_COMPILE__) && __has_builtin(__builtin_amdgcn_global_load_async_to_lds_b128)
#define USE_ASYNC_LDS 1
#else
#define USE_ASYNC_LDS 0
#endif

typedef int v4i_t __attribute__((vector_size(16)));
typedef __attribute__((address_space(1))) v4i_t gv4i_t;   // global v4i
typedef __attribute__((address_space(3))) v4i_t lv4i_t;   // LDS v4i

__device__ __forceinline__ void copy16_to_lds(const void* g, void* l) {
#if USE_ASYNC_LDS
  __builtin_amdgcn_global_load_async_to_lds_b128((gv4i_t*)g, (lv4i_t*)l, 0, 0);
#else
  *reinterpret_cast<U16B*>(l) = *reinterpret_cast<const U16B*>(g);
#endif
}

__device__ __forceinline__ void async_wait0() {
#if USE_ASYNC_LDS
#if __has_builtin(__builtin_amdgcn_s_wait_asynccnt)
  __builtin_amdgcn_s_wait_asynccnt(0);
#else
  asm volatile("s_wait_asynccnt 0x0" ::: "memory");
#endif
#endif
}

// ---------------- weight cast: dst[r*dst_ld + dst_off + c] = bf16(src[r*cols + c])
__global__ void k_castrows(const float* __restrict__ src, bf16_t* __restrict__ dst,
                           int cols, int dst_ld, int dst_off, int total) {
  int t = blockIdx.x * blockDim.x + threadIdx.x;
  if (t >= total) return;
  int r = t / cols, c = t - r * cols;
  dst[(size_t)r * dst_ld + dst_off + c] = f2bf(src[(size_t)r * cols + c]);
}

// ---------------- combined conv bias: 0.25*(b0+b1+b2+b3)
__global__ void k_cbias(const float* b0, const float* b1, const float* b2,
                        const float* b3, float* cb) {
  int t = blockIdx.x * blockDim.x + threadIdx.x;
  if (t < 1024) cb[t] = 0.25f * (b0[t] + b1[t] + b2[t] + b3[t]);
}

// ---------------- adaptive avg pool HxH -> 7x7, write bf16 token matrix (6272 x 3840)
__global__ void k_pool(const float* __restrict__ f, bf16_t* __restrict__ pooled,
                       int cin, int H, int koff, int total) {
  int t = blockIdx.x * blockDim.x + threadIdx.x;
  if (t >= total) return;
  int pos = t % 49;
  int bc  = t / 49;
  int c   = bc % cin;
  int b   = bc / cin;
  int r   = H / 7;
  int h7 = pos / 7, w7 = pos % 7;
  const float* base = f + (((size_t)b * cin + c) * H + (size_t)h7 * r) * H + (size_t)w7 * r;
  float s = 0.f;
  for (int i = 0; i < r; ++i) {
    const float* rowp = base + (size_t)i * H;
    for (int j = 0; j < r; ++j) s += rowp[j];
  }
  pooled[(size_t)(b * 49 + pos) * 3840 + koff + c] = f2bf(s / (float)(r * r));
}

// ---------------- LayerNorm over C=1024, bf16 out
__global__ __launch_bounds__(256) void k_layernorm(const float* __restrict__ x,
    const float* __restrict__ g, const float* __restrict__ bb, bf16_t* __restrict__ out) {
  __shared__ float s1[256], s2[256];
  int row = blockIdx.x, tid = threadIdx.x;
  const float* xr = x + (size_t)row * 1024;
  float v[4], a = 0.f, a2 = 0.f;
#pragma unroll
  for (int i = 0; i < 4; ++i) { v[i] = xr[tid + 256 * i]; a += v[i]; a2 += v[i] * v[i]; }
  s1[tid] = a; s2[tid] = a2; __syncthreads();
  for (int st = 128; st > 0; st >>= 1) {
    if (tid < st) { s1[tid] += s1[tid + st]; s2[tid] += s2[tid + st]; }
    __syncthreads();
  }
  float mean = s1[0] * (1.f / 1024.f);
  float var  = s2[0] * (1.f / 1024.f) - mean * mean;
  float inv  = rsqrtf(var + 1e-5f);
  bf16_t* orow = out + (size_t)row * 1024;
#pragma unroll
  for (int i = 0; i < 4; ++i) {
    int c = tid + 256 * i;
    orow[c] = f2bf((v[i] - mean) * inv * g[c] + bb[c]);
  }
}

// ---------------- WMMA bf16 GEMM: out[m,n] = alpha * sum_k A[m,k]*W[n,k] + bias[n]
// Double-buffered LDS, async global->LDS staging when available.
// mode: 0 = f32 store, 1 = f32 residual accumulate, 2 = GELU -> bf16, 3 = bf16 store
// M % 64 == 0, N % 128 == 0, K % 32 == 0 (guaranteed by caller)
__global__ __launch_bounds__(256) void k_gemm(
    const bf16_t* __restrict__ A, const bf16_t* __restrict__ W,
    const float* __restrict__ bias, float* __restrict__ outf,
    bf16_t* __restrict__ outb, int M, int N, int K, float alpha, int mode) {
  __shared__ __align__(16) bf16_t As[2][64][40];    // 64 x 32 tiles, padded row = 80B
  __shared__ __align__(16) bf16_t Bs[2][128][40];   // 128 x 32 tiles (n-major), padded
  int tid  = threadIdx.x;
  int lane = tid & 31, wv = tid >> 5;
  int wm = wv >> 2, wn = wv & 3;                    // 2x4 wave grid -> 64x128 block tile
  int l15 = lane & 15, hf2 = lane >> 4;
  size_t mb = (size_t)blockIdx.y * 64;
  size_t nb = (size_t)blockIdx.x * 128;
  v8f acc[2][2];
#pragma unroll
  for (int i = 0; i < 2; ++i)
#pragma unroll
    for (int j = 0; j < 2; ++j) acc[i][j] = v8f_zero();

  int arow = tid >> 2, aseg = tid & 3;              // A: 64 rows x 4 x 8 bf16
  int bn = tid >> 1, bh = tid & 1;                  // B: 128 rows x 2 x 16 bf16
  const bf16_t* agp = A + (mb + arow) * (size_t)K + aseg * 8;
  const bf16_t* bgp = W + (nb + bn) * (size_t)K + bh * 16;

  // stage chunk 0 (3 x 16B per thread: 1 for A, 2 for B)
  {
    copy16_to_lds(agp, &As[0][arow][aseg * 8]);
    copy16_to_lds(bgp, &Bs[0][bn][bh * 16]);
    copy16_to_lds(bgp + 8, &Bs[0][bn][bh * 16 + 8]);
  }

  int nchunks = K >> 5;
  for (int c = 0; c < nchunks; ++c) {
    int cur = c & 1;
    async_wait0();        // own chunk-c async writes landed (no-op in sync mode)
    __syncthreads();      // all waves' chunk-c data visible; prior reads of other buf done
    if (c + 1 < nchunks) {
      int kk = (c + 1) << 5;
      copy16_to_lds(agp + kk, &As[1 - cur][arow][aseg * 8]);
      copy16_to_lds(bgp + kk, &Bs[1 - cur][bn][bh * 16]);
      copy16_to_lds(bgp + kk + 8, &Bs[1 - cur][bn][bh * 16 + 8]);
    }

    FragU af[2];
#pragma unroll
    for (int i = 0; i < 2; ++i) {
      int ml = wm * 32 + i * 16 + l15;
      af[i].q[0] = *reinterpret_cast<const U16B*>(&As[cur][ml][hf2 * 8]);       // K = half*8 + 0..7
      af[i].q[1] = *reinterpret_cast<const U16B*>(&As[cur][ml][16 + hf2 * 8]);  // K = 16 + half*8 + 0..7
    }
#pragma unroll
    for (int j = 0; j < 2; ++j) {
      int nl = wn * 32 + j * 16 + l15;
      FragU bu;
      bu.q[0] = *reinterpret_cast<const U16B*>(&Bs[cur][nl][hf2 * 8]);          // K = half*8 + 0..7
      bu.q[1] = *reinterpret_cast<const U16B*>(&Bs[cur][nl][16 + hf2 * 8]);     // K = 16 + half*8 + 0..7
      acc[0][j] = __builtin_amdgcn_wmma_f32_16x16x32_bf16(false, af[0].v, false, bu.v,
                                                          (short)0, acc[0][j], false, false);
      acc[1][j] = __builtin_amdgcn_wmma_f32_16x16x32_bf16(false, af[1].v, false, bu.v,
                                                          (short)0, acc[1][j], false, false);
    }
  }

#pragma unroll
  for (int i = 0; i < 2; ++i)
#pragma unroll
    for (int j = 0; j < 2; ++j)
#pragma unroll
      for (int vg = 0; vg < 8; ++vg) {
        size_t m = mb + wm * 32 + i * 16 + vg + hf2 * 8;   // C layout: VGPR vg -> M=vg(+8 hi half)
        size_t n = nb + wn * 32 + j * 16 + l15;
        float v = alpha * acc[i][j][vg] + bias[n];
        size_t idx = m * (size_t)N + n;
        if (mode == 0)      outf[idx] = v;
        else if (mode == 1) outf[idx] += v;
        else if (mode == 2) outb[idx] = f2bf(0.5f * v * (1.0f + erff(v * 0.70710678118654752f)));
        else                outb[idx] = f2bf(v);
      }
}

// ---------------- window attention: one block = one (batch, head); 4 waves
__global__ __launch_bounds__(128) void k_attn(const float* __restrict__ qkv,
    const float* __restrict__ rpb, bf16_t* __restrict__ o) {
  __shared__ __align__(16) bf16_t Qb[64][72];
  __shared__ __align__(16) bf16_t Kb[64][72];
  __shared__ __align__(16) bf16_t Vt[64][72];   // transposed: Vt[d][n]
  __shared__ float Sb[64][65];
  int tid  = threadIdx.x;
  int b    = blockIdx.x >> 4;
  int head = blockIdx.x & 15;

  for (int idx = tid; idx < 4096; idx += 128) {
    int n = idx >> 6, d = idx & 63;
    float q = 0.f, k = 0.f, vv = 0.f;
    if (n < 49) {
      size_t base = (size_t)(b * 49 + n) * 3072;
      q  = qkv[base + head * 64 + d] * 0.125f;          // scale = 64^-0.5
      k  = qkv[base + 1024 + head * 64 + d];
      vv = qkv[base + 2048 + head * 64 + d];
    }
    Qb[n][d] = f2bf(q);
    Kb[n][d] = f2bf(k);
    Vt[d][n] = f2bf(vv);
  }
  __syncthreads();

  int lane = tid & 31, wv = tid >> 5;
  int l15 = lane & 15, hf2 = lane >> 4;

  // S = Q * K^T  (each wave: one 16-row M tile, all four 16-col N tiles)
  {
    v8f acc[4];
#pragma unroll
    for (int j = 0; j < 4; ++j) acc[j] = v8f_zero();
#pragma unroll
    for (int kk = 0; kk < 64; kk += 32) {
      FragU a;
      int ml = wv * 16 + l15;
      a.q[0] = *reinterpret_cast<const U16B*>(&Qb[ml][kk + hf2 * 8]);
      a.q[1] = *reinterpret_cast<const U16B*>(&Qb[ml][kk + 16 + hf2 * 8]);
#pragma unroll
      for (int j = 0; j < 4; ++j) {
        int nl = j * 16 + l15;
        FragU bu;
        bu.q[0] = *reinterpret_cast<const U16B*>(&Kb[nl][kk + hf2 * 8]);
        bu.q[1] = *reinterpret_cast<const U16B*>(&Kb[nl][kk + 16 + hf2 * 8]);
        acc[j] = __builtin_amdgcn_wmma_f32_16x16x32_bf16(false, a.v, false, bu.v,
                                                         (short)0, acc[j], false, false);
      }
    }
#pragma unroll
    for (int j = 0; j < 4; ++j)
#pragma unroll
      for (int vg = 0; vg < 8; ++vg)
        Sb[wv * 16 + vg + hf2 * 8][j * 16 + l15] = acc[j][vg];
  }
  __syncthreads();

  // softmax with relative-position bias (one thread per valid query row)
  if (tid < 49) {
    int r = tid, ry = r / 7, rx = r % 7;
    float mx = -1e30f;
    for (int c2 = 0; c2 < 49; ++c2) {
      int cy = c2 / 7, cx = c2 % 7;
      int ridx = (ry - cy + 6) * 13 + (rx - cx + 6);
      float val = Sb[r][c2] + rpb[(size_t)ridx * 16 + head];
      Sb[r][c2] = val;
      mx = fmaxf(mx, val);
    }
    float s = 0.f;
    for (int c2 = 0; c2 < 49; ++c2) { float e = __expf(Sb[r][c2] - mx); Sb[r][c2] = e; s += e; }
    float inv = 1.f / s;
    for (int c2 = 0; c2 < 49; ++c2) Sb[r][c2] *= inv;
    for (int c2 = 49; c2 < 64; ++c2) Sb[r][c2] = 0.f;
  }
  __syncthreads();

  // O = P * V
  {
    v8f acc[4];
#pragma unroll
    for (int j = 0; j < 4; ++j) acc[j] = v8f_zero();
#pragma unroll
    for (int kk = 0; kk < 64; kk += 32) {
      FragU a;
      int ml = wv * 16 + l15;
#pragma unroll
      for (int e = 0; e < 16; ++e) {
        int kq = kk + ((e < 8) ? (hf2 * 8 + e) : (16 + hf2 * 8 + e - 8));
        a.h[e] = f2bf(Sb[ml][kq]);
      }
#pragma unroll
      for (int j = 0; j < 4; ++j) {
        int dl = j * 16 + l15;
        FragU bu;
        bu.q[0] = *reinterpret_cast<const U16B*>(&Vt[dl][kk + hf2 * 8]);
        bu.q[1] = *reinterpret_cast<const U16B*>(&Vt[dl][kk + 16 + hf2 * 8]);
        acc[j] = __builtin_amdgcn_wmma_f32_16x16x32_bf16(false, a.v, false, bu.v,
                                                         (short)0, acc[j], false, false);
      }
    }
#pragma unroll
    for (int j = 0; j < 4; ++j)
#pragma unroll
      for (int vg = 0; vg < 8; ++vg) {
        int m = wv * 16 + vg + hf2 * 8;
        if (m < 49)
          o[(size_t)(b * 49 + m) * 1024 + head * 64 + j * 16 + l15] = f2bf(acc[j][vg]);
      }
  }
}

// ---------------- final token mean pool: (128,49,1024) -> (128,1024)
__global__ void k_meanpool(const float* __restrict__ x, float* __restrict__ out) {
  int t = blockIdx.x * blockDim.x + threadIdx.x;
  if (t >= 131072) return;
  int b = t >> 10, c = t & 1023;
  const float* p = x + (size_t)b * 49 * 1024 + c;
  float s = 0.f;
  for (int n = 0; n < 49; ++n) s += p[(size_t)n * 1024];
  out[t] = s * (1.f / 49.f);
}

extern "C" void kernel_launch(void* const* d_in, const int* in_sizes, int n_in,
                              void* d_out, int out_size, void* d_ws, size_t ws_size,
                              hipStream_t stream) {
  (void)in_sizes; (void)n_in; (void)out_size; (void)ws_size;
  char* ws = (char*)d_ws;
  size_t off = 0;
  auto take = [&](size_t bytes) -> char* {
    char* p = ws + off;
    off += (bytes + 255) & ~(size_t)255;
    return p;
  };
  bf16_t* Wc     = (bf16_t*)take(1024ull * 3840 * 2);
  bf16_t* Wqkv   = (bf16_t*)take(2ull * 3072 * 1024 * 2);
  bf16_t* Wproj  = (bf16_t*)take(2ull * 1024 * 1024 * 2);
  bf16_t* Wfc1   = (bf16_t*)take(2ull * 4096 * 1024 * 2);
  bf16_t* Wfc2   = (bf16_t*)take(2ull * 1024 * 4096 * 2);
  float*  cbias  = (float*)take(1024 * 4);
  bf16_t* pooled = (bf16_t*)take(6272ull * 3840 * 2);
  float*  xtok   = (float*)take(6272ull * 1024 * 4);
  bf16_t* h1     = (bf16_t*)take(6272ull * 1024 * 2);
  bf16_t* h2     = (bf16_t*)take(6272ull * 4096 * 2);
  float*  qkvb   = (float*)take(6272ull * 3072 * 4);

  auto cdiv = [](int a, int b) { return (a + b - 1) / b; };

  // weight casts (conv weights concatenated into one 1024x3840 matrix)
  {
    int cins[4] = {256, 512, 1024, 2048};
    int koff = 0;
    for (int s = 0; s < 4; ++s) {
      int total = 1024 * cins[s];
      k_castrows<<<cdiv(total, 256), 256, 0, stream>>>(
          (const float*)d_in[4 + 2 * s], Wc, cins[s], 3840, koff, total);
      koff += cins[s];
    }
    k_castrows<<<cdiv(2 * 3072 * 1024, 256), 256, 0, stream>>>(
        (const float*)d_in[14], Wqkv, 1024, 1024, 0, 2 * 3072 * 1024);
    k_castrows<<<cdiv(2 * 1024 * 1024, 256), 256, 0, stream>>>(
        (const float*)d_in[17], Wproj, 1024, 1024, 0, 2 * 1024 * 1024);
    k_castrows<<<cdiv(2 * 4096 * 1024, 256), 256, 0, stream>>>(
        (const float*)d_in[21], Wfc1, 1024, 1024, 0, 2 * 4096 * 1024);
    k_castrows<<<cdiv(2 * 1024 * 4096, 256), 256, 0, stream>>>(
        (const float*)d_in[23], Wfc2, 4096, 4096, 0, 2 * 1024 * 4096);
  }
  k_cbias<<<4, 256, 0, stream>>>((const float*)d_in[5], (const float*)d_in[7],
                                 (const float*)d_in[9], (const float*)d_in[11], cbias);

  // pooling (memory-bound stage)
  {
    int cins[4] = {256, 512, 1024, 2048};
    int Hs[4]   = {56, 28, 14, 7};
    int koff = 0;
    for (int s = 0; s < 4; ++s) {
      int total = 128 * cins[s] * 49;
      k_pool<<<cdiv(total, 256), 256, 0, stream>>>(
          (const float*)d_in[s], pooled, cins[s], Hs[s], koff, total);
      koff += cins[s];
    }
  }

  // fused 1x1 conv over all scales, averaged: xtok = 0.25 * pooled @ Wc^T + cbias
  k_gemm<<<dim3(1024 / 128, 6272 / 64), 256, 0, stream>>>(
      pooled, Wc, cbias, xtok, nullptr, 6272, 1024, 3840, 0.25f, 0);

  for (int i = 0; i < 2; ++i) {
    const float* ln1g  = (const float*)d_in[12] + i * 1024;
    const float* ln1b  = (const float*)d_in[13] + i * 1024;
    const float* bqkv  = (const float*)d_in[15] + i * 3072;
    const float* rpbp  = (const float*)d_in[16] + (size_t)i * 169 * 16;
    const float* bproj = (const float*)d_in[18] + i * 1024;
    const float* ln2g  = (const float*)d_in[19] + i * 1024;
    const float* ln2b  = (const float*)d_in[20] + i * 1024;
    const float* bfc1  = (const float*)d_in[22] + i * 4096;
    const float* bfc2  = (const float*)d_in[24] + i * 1024;

    k_layernorm<<<6272, 256, 0, stream>>>(xtok, ln1g, ln1b, h1);
    k_gemm<<<dim3(3072 / 128, 98), 256, 0, stream>>>(
        h1, Wqkv + (size_t)i * 3072 * 1024, bqkv, qkvb, nullptr, 6272, 3072, 1024, 1.f, 0);
    k_attn<<<2048, 128, 0, stream>>>(qkvb, rpbp, h1);
    k_gemm<<<dim3(1024 / 128, 98), 256, 0, stream>>>(
        h1, Wproj + (size_t)i * 1024 * 1024, bproj, xtok, nullptr, 6272, 1024, 1024, 1.f, 1);
    k_layernorm<<<6272, 256, 0, stream>>>(xtok, ln2g, ln2b, h1);
    k_gemm<<<dim3(4096 / 128, 98), 256, 0, stream>>>(
        h1, Wfc1 + (size_t)i * 4096 * 1024, bfc1, nullptr, h2, 6272, 4096, 1024, 1.f, 2);
    k_gemm<<<dim3(1024 / 128, 98), 256, 0, stream>>>(
        h2, Wfc2 + (size_t)i * 1024 * 4096, bfc2, xtok, nullptr, 6272, 1024, 4096, 1.f, 1);
  }

  k_meanpool<<<512, 256, 0, stream>>>(xtok, (float*)d_out);
}
